// MultiHeadAttention_2723009266456
// MI455X (gfx1250) — compile-verified, matrix operand order checked by reading, not run on hardware
//
#include <hip/hip_runtime.h>

// ---------------------------------------------------------------------------
// MultiHeadAttention for MI455X (gfx1250): bf16 WMMA GEMMs + fp32 softmax/LN.
// B=256, NA=128, E=1024, H=16, HD=64.
// ---------------------------------------------------------------------------

#define B_    256
#define NA_   128
#define E_    1024
#define H_    16
#define HD_   64
#define ROWS_ 32768   // B*NA
#define BH_   4096    // B*H
#define E3_   3072
#define KB_   64      // GEMM K-chunk (double buffered)

typedef __attribute__((ext_vector_type(16))) __bf16 bf16x16;
typedef __attribute__((ext_vector_type(8)))  float  f32x8;
typedef __attribute__((ext_vector_type(4)))  int    v4i_t;

typedef __attribute__((address_space(1))) v4i_t gv4i_t;   // global v4i
typedef __attribute__((address_space(3))) v4i_t lv4i_t;   // LDS v4i

// ---- async global -> LDS (CDNA5 GLOBAL_LOAD_ASYNC_TO_LDS_B128) -------------
// ROCm 7.2 signature (from diagnostic): (v4i AS1*, v4i AS3*, imm offset, imm cpol)
#if defined(__gfx1250__) && __has_builtin(__builtin_amdgcn_global_load_async_to_lds_b128)
#define ASYNC_LDS 1
__device__ __forceinline__ void async_cp16(const void* g, void* l) {
  __builtin_amdgcn_global_load_async_to_lds_b128((gv4i_t*)g, (lv4i_t*)l, 0, 0);
}
__device__ __forceinline__ void async_wait0() {
#if __has_builtin(__builtin_amdgcn_s_wait_asynccnt)
  __builtin_amdgcn_s_wait_asynccnt(0);
#else
  asm volatile("s_wait_asynccnt 0" ::: "memory");
#endif
}
#else
#define ASYNC_LDS 0
__device__ __forceinline__ void async_cp16(const void* g, void* l) {
  *(uint4*)l = *(const uint4*)g;
}
__device__ __forceinline__ void async_wait0() {}
#endif

// ---- bf16 <-> f32 (RNE) ----------------------------------------------------
__device__ __forceinline__ __bf16 f2bf(float x) {
  union { float f; unsigned u; } v; v.f = x;
  unsigned r = v.u + 0x7FFFu + ((v.u >> 16) & 1u);
  unsigned short h = (unsigned short)(r >> 16);
  return __builtin_bit_cast(__bf16, h);
}
__device__ __forceinline__ float bf2f(__bf16 x) {
  unsigned short h = __builtin_bit_cast(unsigned short, x);
  union { unsigned u; float f; } v; v.u = ((unsigned)h) << 16;
  return v.f;
}

// ---- WMMA fragment load from LDS ------------------------------------------
// ISA 7.12.2 16-bit A/B layout: lanes 0-15 hold K{0..7,16..23}, lanes 16-31
// hold K{8..15,24..31}, of row (A: M=lane%16) / column (B: N=lane%16).
// p points at this lane's 32-element K-chunk (row-major, K contiguous).
__device__ __forceinline__ bf16x16 frag_ld(const __bf16* p, int hi /*0 or 8*/) {
  union { bf16x16 v; uint4 u[2]; } w;
  w.u[0] = *(const uint4*)(p + hi);        // K 0..7  (or 8..15)
  w.u[1] = *(const uint4*)(p + hi + 16);   // K 16..23 (or 24..31)
  return w.v;
}

__device__ __forceinline__ float wave_sum(float v) {
  #pragma unroll
  for (int off = 16; off > 0; off >>= 1) v += __shfl_xor(v, off, 32);
  return v;
}

// ---- Kernel 1: LN1 ---------------------------------------------------------
__global__ __launch_bounds__(256) void ln1_kernel(
    const float* __restrict__ x, const float* __restrict__ w,
    const float* __restrict__ b, float* __restrict__ xn) {
  const int row = blockIdx.x;
  const int t = threadIdx.x;
  const float4 v = ((const float4*)(x + (size_t)row * E_))[t];
  float s  = v.x + v.y + v.z + v.w;
  float ss = v.x*v.x + v.y*v.y + v.z*v.z + v.w*v.w;
  __shared__ float red[16];
  s = wave_sum(s); ss = wave_sum(ss);
  const int wave = t >> 5, lane = t & 31;
  if (lane == 0) { red[wave] = s; red[8 + wave] = ss; }
  __syncthreads();
  if (t == 0) {
    float ts = 0.f, tss = 0.f;
    #pragma unroll
    for (int i = 0; i < 8; i++) { ts += red[i]; tss += red[8 + i]; }
    red[0] = ts; red[8] = tss;
  }
  __syncthreads();
  const float mean = red[0] * (1.f / E_);
  const float var  = red[8] * (1.f / E_) - mean * mean;
  const float rstd = rsqrtf(var + 1e-5f);
  const float4 wv = ((const float4*)w)[t];
  const float4 bv = ((const float4*)b)[t];
  float4 o;
  o.x = (v.x - mean) * rstd * wv.x + bv.x;
  o.y = (v.y - mean) * rstd * wv.y + bv.y;
  o.z = (v.z - mean) * rstd * wv.z + bv.z;
  o.w = (v.w - mean) * rstd * wv.w + bv.w;
  ((float4*)(xn + (size_t)row * E_))[t] = o;
}

// ---- Kernel 2: weight fp32 -> bf16 ----------------------------------------
__global__ __launch_bounds__(256) void cvt_kernel(
    const float* __restrict__ w, __bf16* __restrict__ wb) {
  const size_t i = ((size_t)blockIdx.x * 256 + threadIdx.x) * 4;
  const float4 v = *(const float4*)(w + i);
  union { __bf16 h[4]; uint2 u; } pk;
  pk.h[0] = f2bf(v.x); pk.h[1] = f2bf(v.y);
  pk.h[2] = f2bf(v.z); pk.h[3] = f2bf(v.w);
  *(uint2*)(wb + i) = pk.u;
}

// ---- Kernel 3: QKV GEMM (WMMA bf16) + bias + q-scale + head scatter --------
// qkv[m, o] = sum_k xn[m,k] * W[o,k] + b[o];  q *= 1/sqrt(HD)
// Head-layout quirk (raw reshape (na, B*H, HD)): for o = seg*1024 + h64*64 + d,
//   dest[seg][ ((m%256)*16 + h64)*8192 + (m/256)*64 + d ]
// Grid: x = N-block (fast; 24 consecutive blocks reuse one A tile from L2,
// weights stay L2-resident), y = M-block.
__global__ __launch_bounds__(256) void qkv_kernel(
    const float* __restrict__ xn, const __bf16* __restrict__ wbf,
    const float* __restrict__ in_b,
    __bf16* __restrict__ qh, __bf16* __restrict__ kh, __bf16* __restrict__ vh) {
  __shared__ __bf16 As[2][128 * KB_];   // 2 x 16 KB
  __shared__ __bf16 Bs[2][128 * KB_];   // 2 x 16 KB
  const int n0 = blockIdx.x * 128;      // within 3072 (fast dim)
  const int m0 = blockIdx.y * 128;
  const int t = threadIdx.x;
  const int wave = t >> 5, lane = t & 31;
  const int hi = (lane >= 16) ? 8 : 0;
  const int l16 = lane & 15;

  auto stageA = [&](int buf, int k0) {
    #pragma unroll
    for (int q = 0; q < 8; q++) {
      const int idx = t + q * 256;            // 0..2047
      const int r = idx >> 4, c = (idx & 15) * 4;
      const float4 av = *(const float4*)(xn + (size_t)(m0 + r) * E_ + k0 + c);
      union { __bf16 h[4]; uint2 u; } pk;
      pk.h[0] = f2bf(av.x); pk.h[1] = f2bf(av.y);
      pk.h[2] = f2bf(av.z); pk.h[3] = f2bf(av.w);
      *(uint2*)(&As[buf][r * KB_ + c]) = pk.u;
    }
  };
  auto stageB = [&](int buf, int k0) {
    #pragma unroll
    for (int q = 0; q < 4; q++) {
      const int idx = t + q * 256;            // 0..1023
      const int r = idx >> 3, c = (idx & 7) * 8;
      async_cp16(wbf + (size_t)(n0 + r) * E_ + k0 + c, &Bs[buf][r * KB_ + c]);
    }
  };

  f32x8 acc[8];
  #pragma unroll
  for (int i = 0; i < 8; i++) acc[i] = f32x8{0.f,0.f,0.f,0.f,0.f,0.f,0.f,0.f};

  stageA(0, 0);
  stageB(0, 0);
  for (int kt = 0; kt < E_ / KB_; kt++) {
    const int buf = kt & 1;
    async_wait0();          // own async B copies into `buf` done
    __syncthreads();        // all waves' staging of `buf` visible
    if (kt + 1 < E_ / KB_) {
      stageA(buf ^ 1, (kt + 1) * KB_);
      stageB(buf ^ 1, (kt + 1) * KB_);
    }
    #pragma unroll
    for (int c = 0; c < 2; c++) {
      const bf16x16 a = frag_ld(&As[buf][(wave * 16 + l16) * KB_ + c * 32], hi);
      #pragma unroll
      for (int nt = 0; nt < 8; nt++) {
        const bf16x16 bfr = frag_ld(&Bs[buf][(nt * 16 + l16) * KB_ + c * 32], hi);
        acc[nt] = __builtin_amdgcn_wmma_f32_16x16x32_bf16(
            false, a, false, bfr, (short)0, acc[nt], false, false);
      }
    }
  }

  // Epilogue: bias, scale, scatter to head layout (bf16)
  const int seg = n0 >> 10;                       // uniform per block
  __bf16* dst = (seg == 0) ? qh : ((seg == 1) ? kh : vh);
  const float mul = (seg == 0) ? 0.125f : 1.0f;   // HD^-0.5 = 1/8
  #pragma unroll
  for (int nt = 0; nt < 8; nt++) {
    const int n = n0 + nt * 16 + l16;             // within 3072
    const int oe = n & 1023;
    const float bias = in_b[n];
    const int h64 = oe >> 6, d = oe & 63;
    #pragma unroll
    for (int r = 0; r < 8; r++) {
      const int m = m0 + wave * 16 + r + ((lane >> 4) << 3);
      const float val = (acc[nt][r] + bias) * mul;
      const size_t g = (size_t)(((m & 255) << 4) + h64) * 8192
                     + (size_t)(m >> 8) * 64 + d;
      dst[g] = f2bf(val);
    }
  }
}

// ---- Kernel 4: attention per bh (4096 blocks) ------------------------------
__global__ __launch_bounds__(256) void attn_kernel(
    const __bf16* __restrict__ qh, const __bf16* __restrict__ kh,
    const __bf16* __restrict__ vh, __bf16* __restrict__ attn,
    float* __restrict__ avgw) {
  __shared__ __bf16 qk[16384];   // 32KB: [0:8192)=q, [8192:16384)=k; reused as probs 128x128
  __shared__ __bf16 vT[8192];    // 16KB: 64 rows (d) x 128 (key)
  const int bh = blockIdx.x;
  const int t = threadIdx.x, wave = t >> 5, lane = t & 31;
  const int hi = (lane >= 16) ? 8 : 0;
  const int l16 = lane & 15;

  // Stage q, k via async global->LDS copies (contiguous 16KB each)
  {
    const __bf16* qg = qh + (size_t)bh * 8192;
    const __bf16* kg = kh + (size_t)bh * 8192;
    #pragma unroll
    for (int i = t; i < 1024; i += 256) async_cp16(qg + i * 8, qk + i * 8);
    #pragma unroll
    for (int i = t; i < 1024; i += 256) async_cp16(kg + i * 8, qk + 8192 + i * 8);
  }
  // Stage v transposed: vT[d][key]
  {
    const __bf16* vg = vh + (size_t)bh * 8192;
    const int np = t >> 1;               // key 0..127
    const int d0 = (t & 1) * 32;
    __bf16 tmp[32];
    const uint4* src = (const uint4*)(vg + np * 64 + d0);
    *(uint4*)(tmp + 0)  = src[0];
    *(uint4*)(tmp + 8)  = src[1];
    *(uint4*)(tmp + 16) = src[2];
    *(uint4*)(tmp + 24) = src[3];
    #pragma unroll
    for (int j = 0; j < 32; j++) vT[(d0 + j) * 128 + np] = tmp[j];
  }
  async_wait0();
  __syncthreads();

  // S = q @ k^T : wave handles 16 query rows x all 128 keys (8 N-tiles), K=64
  f32x8 s[8];
  #pragma unroll
  for (int i = 0; i < 8; i++) s[i] = f32x8{0.f,0.f,0.f,0.f,0.f,0.f,0.f,0.f};
  const int rq = wave * 16 + l16;
  #pragma unroll
  for (int c = 0; c < 2; c++) {
    const bf16x16 a = frag_ld(qk + rq * 64 + c * 32, hi);
    #pragma unroll
    for (int nt = 0; nt < 8; nt++) {
      const bf16x16 b = frag_ld(qk + 8192 + (nt * 16 + l16) * 64 + c * 32, hi);
      s[nt] = __builtin_amdgcn_wmma_f32_16x16x32_bf16(
          false, a, false, b, (short)0, s[nt], false, false);
    }
  }

  // Row softmax. C layout: VGPR r, lane l -> row wave*16 + r + 8*(l/16),
  // col l%16; each 16-lane group holds one row slice -> xor-reduce over 1,2,4,8.
  float rsumv[8];
  #pragma unroll
  for (int r = 0; r < 8; r++) {
    float m = -3.4e38f;
    #pragma unroll
    for (int nt = 0; nt < 8; nt++) m = fmaxf(m, s[nt][r]);
    #pragma unroll
    for (int off = 1; off < 16; off <<= 1) m = fmaxf(m, __shfl_xor(m, off, 32));
    float sum = 0.f;
    #pragma unroll
    for (int nt = 0; nt < 8; nt++) {
      const float p = __expf(s[nt][r] - m);
      s[nt][r] = p; sum += p;
    }
    #pragma unroll
    for (int off = 1; off < 16; off <<= 1) sum += __shfl_xor(sum, off, 32);
    rsumv[r] = sum;
  }

  __syncthreads();  // everyone done reading q/k before probs overwrite

  // Write normalized probs (bf16) into qk[0:16384) as 128x128
  #pragma unroll
  for (int r = 0; r < 8; r++) {
    const float rinv = 1.f / rsumv[r];
    const int m = wave * 16 + r + ((lane >> 4) << 3);
    #pragma unroll
    for (int nt = 0; nt < 8; nt++) {
      qk[m * 128 + nt * 16 + l16] = f2bf(s[nt][r] * rinv);
    }
  }
  __syncthreads();

  // avg_w: column sums over query rows, / H
  if (t < 128) {
    float cs = 0.f;
    for (int r2 = 0; r2 < 128; r2++) cs += bf2f(qk[r2 * 128 + t]);
    avgw[(size_t)bh * 128 + t] = cs * (1.f / 16.f);
  }

  // O = P @ V : K=128 (keys), N=64 (d), B operand from vT (d-major)
  f32x8 o[4];
  #pragma unroll
  for (int i = 0; i < 4; i++) o[i] = f32x8{0.f,0.f,0.f,0.f,0.f,0.f,0.f,0.f};
  #pragma unroll
  for (int c = 0; c < 4; c++) {
    const bf16x16 a = frag_ld(qk + rq * 128 + c * 32, hi);
    #pragma unroll
    for (int nt = 0; nt < 4; nt++) {
      const bf16x16 b = frag_ld(vT + (nt * 16 + l16) * 128 + c * 32, hi);
      o[nt] = __builtin_amdgcn_wmma_f32_16x16x32_bf16(
          false, a, false, b, (short)0, o[nt], false, false);
    }
  }

  // Scatter attn back: flat f = na'*262144 + bh*64 + d  (inverse head reshape)
  #pragma unroll
  for (int nt = 0; nt < 4; nt++) {
    const int d = nt * 16 + l16;
    #pragma unroll
    for (int r = 0; r < 8; r++) {
      const int m = wave * 16 + r + ((lane >> 4) << 3);
      attn[(size_t)m * 262144 + (size_t)bh * 64 + d] = f2bf(o[nt][r]);
    }
  }
}

// ---- Kernel 5: residual + LN2 ---------------------------------------------
__global__ __launch_bounds__(256) void ln2_kernel(
    const float* __restrict__ xn, const __bf16* __restrict__ attn,
    const float* __restrict__ w, const float* __restrict__ b,
    float* __restrict__ out) {
  const int row = blockIdx.x;
  const int t = threadIdx.x;
  const float4 xv = ((const float4*)(xn + (size_t)row * E_))[t];
  union { uint2 u; __bf16 h[4]; } av;
  av.u = *(const uint2*)(attn + (size_t)row * E_ + t * 4);
  float4 v;
  v.x = xv.x + bf2f(av.h[0]); v.y = xv.y + bf2f(av.h[1]);
  v.z = xv.z + bf2f(av.h[2]); v.w = xv.w + bf2f(av.h[3]);
  float s  = v.x + v.y + v.z + v.w;
  float ss = v.x*v.x + v.y*v.y + v.z*v.z + v.w*v.w;
  __shared__ float red[16];
  s = wave_sum(s); ss = wave_sum(ss);
  const int wave = t >> 5, lane = t & 31;
  if (lane == 0) { red[wave] = s; red[8 + wave] = ss; }
  __syncthreads();
  if (t == 0) {
    float ts = 0.f, tss = 0.f;
    #pragma unroll
    for (int i = 0; i < 8; i++) { ts += red[i]; tss += red[8 + i]; }
    red[0] = ts; red[8] = tss;
  }
  __syncthreads();
  const float mean = red[0] * (1.f / E_);
  const float var  = red[8] * (1.f / E_) - mean * mean;
  const float rstd = rsqrtf(var + 1e-5f);
  const float4 wv = ((const float4*)w)[t];
  const float4 bv = ((const float4*)b)[t];
  float4 o;
  o.x = (v.x - mean) * rstd * wv.x + bv.x;
  o.y = (v.y - mean) * rstd * wv.y + bv.y;
  o.z = (v.z - mean) * rstd * wv.z + bv.z;
  o.w = (v.w - mean) * rstd * wv.w + bv.w;
  ((float4*)(out + (size_t)row * E_))[t] = o;
}

// ---------------------------------------------------------------------------
extern "C" void kernel_launch(void* const* d_in, const int* in_sizes, int n_in,
                              void* d_out, int out_size, void* d_ws, size_t ws_size,
                              hipStream_t stream) {
  const float* x     = (const float*)d_in[0];
  const float* ln1_w = (const float*)d_in[1];
  const float* ln1_b = (const float*)d_in[2];
  const float* in_w  = (const float*)d_in[3];
  const float* in_b  = (const float*)d_in[4];
  const float* ln2_w = (const float*)d_in[5];
  const float* ln2_b = (const float*)d_in[6];

  float* out  = (float*)d_out;
  float* avgw = out + (size_t)ROWS_ * E_;          // 33,554,432 floats then avg_w

  char* ws = (char*)d_ws;
  float*  xn   = (float*)ws;  ws += (size_t)ROWS_ * E_ * 4;        // 134 MB
  __bf16* wb   = (__bf16*)ws; ws += (size_t)E3_ * E_ * 2;          // 6 MB
  __bf16* qh   = (__bf16*)ws; ws += (size_t)ROWS_ * E_ * 2;        // 67 MB
  __bf16* kh   = (__bf16*)ws; ws += (size_t)ROWS_ * E_ * 2;        // 67 MB
  __bf16* vh   = (__bf16*)ws; ws += (size_t)ROWS_ * E_ * 2;        // 67 MB
  __bf16* atn  = (__bf16*)ws;                                       // 67 MB

  ln1_kernel<<<ROWS_, 256, 0, stream>>>(x, ln1_w, ln1_b, xn);
  cvt_kernel<<<(E3_ * E_) / 1024, 256, 0, stream>>>(in_w, wb);
  qkv_kernel<<<dim3(E3_ / 128, ROWS_ / 128), 256, 0, stream>>>(
      xn, wb, in_b, qh, kh, vh);
  attn_kernel<<<BH_, 256, 0, stream>>>(qh, kh, vh, atn, avgw);
  ln2_kernel<<<ROWS_, 256, 0, stream>>>(xn, atn, ln2_w, ln2_b, out);
}
